// Gemma2Attention_18769007084653
// MI455X (gfx1250) — compile-verified, matrix operand order checked by reading, not run on hardware
//
#include <hip/hip_runtime.h>
#include <math.h>

#define DEV __device__ __forceinline__

typedef __bf16 bf16;
typedef __attribute__((ext_vector_type(16))) __bf16 v16bf;
typedef __attribute__((ext_vector_type(8)))  float   v8f;
typedef __attribute__((ext_vector_type(4)))  unsigned int u32x4;

union BFrag {
  v16bf v;
  u32x4 u4[2];
};

static constexpr int   Bn   = 2;
static constexpr int   Sn   = 2048;
static constexpr int   Dn   = 2048;
static constexpr int   NQn  = 16;
static constexpr int   NKVn = 8;
static constexpr int   Hn   = 256;
static constexpr int   Gn   = 2;
static constexpr int   WIN  = 1024;
static constexpr float CAP  = 50.0f;
static constexpr float LOG1E4 = 9.2103403719761836f; // ln(10000)

DEV v8f v8f_zero() { v8f z = {0.f,0.f,0.f,0.f,0.f,0.f,0.f,0.f}; return z; }

DEV v8f wmma_bf16(const BFrag& a, const BFrag& b, v8f c) {
  // D = A(16x32 bf16) * B(32x16 bf16) + C(16x16 f32)
  return __builtin_amdgcn_wmma_f32_16x16x32_bf16(
      false, a.v, false, b.v, (short)0, c, false, false);
}

// fast tanh via exp2-based __expf; clamp keeps e^{2x} finite (tanh saturated)
DEV float fast_tanh(float x) {
  x = fminf(fmaxf(x, -15.f), 15.f);
  float e = __expf(2.f * x);
  return (e - 1.f) / (e + 1.f);
}

// ---------------------------------------------------------------------------
// Generic tiled GEMM: C[M,N] = A[M,K] * B[K,N]
// Block tile: 128(M) x 256(N) x 32(K), 256 threads = 8 waves,
// wave w computes rows [w*16, w*16+16) x all 256 cols (16 WMMA accumulators).
// ROPE epilogue uses the permutation-equivalent form (pairs at (i, i+128)):
// valid because logits = q.kT is invariant to a shared h-permutation of q,k.
// ---------------------------------------------------------------------------
template<bool ABF16, typename CT, bool ROPE>
__global__ __launch_bounds__(256)
void gemm_kernel(const void* __restrict__ Abase, const float* __restrict__ Bbase,
                 CT* __restrict__ Cbase, int K,
                 int ldA, int ldB, int ldC,
                 int NH, long Abatch, long Bhead, long Cjob, float outScale)
{
  __shared__ bf16 As[128][40];   // A tile (row-major), 80B row stride
  __shared__ bf16 Bs[256][40];   // B tile transposed to [n][k], 80B row stride

  const int job  = blockIdx.z;
  const int b    = job / NH;
  const int head = job % NH;
  const float* Af = ABF16 ? nullptr : ((const float*)Abase + (long)b * Abatch);
  const bf16*  Ab = ABF16 ? ((const bf16*)Abase + (long)b * Abatch) : nullptr;
  const float* Bp = Bbase + (long)head * Bhead;
  CT*          Cp = Cbase + (long)job * Cjob;

  const int m0   = blockIdx.y * 128;
  const int n0   = blockIdx.x * 256;
  const int tid  = threadIdx.x;
  const int wave = tid >> 5;
  const int lane = tid & 31;
  const int l16  = lane & 15;
  const int lh   = lane >> 4;   // 0 for lanes 0-15, 1 for lanes 16-31

  v8f acc[16];
#pragma unroll
  for (int j = 0; j < 16; ++j) acc[j] = v8f_zero();

  // staging coordinates
  const int arow = tid >> 1;          // A: thread -> (row, 16-col segment)
  const int aseg = tid & 1;

  for (int k0 = 0; k0 < K; k0 += 32) {
    // ---- stage A tile 128x32: vector loads, packed b128 LDS stores ----
    {
      union { bf16 h[16]; u32x4 q[2]; } tmp;
      if (ABF16) {
        const bf16* sb = Ab + (long)(m0 + arow) * ldA + k0 + aseg * 16;
        tmp.q[0] = *(const u32x4*)(sb);
        tmp.q[1] = *(const u32x4*)(sb + 8);
      } else {
        const float* src = Af + (long)(m0 + arow) * ldA + k0 + aseg * 16;
#pragma unroll
        for (int i = 0; i < 16; ++i) tmp.h[i] = (bf16)src[i];
      }
      *(u32x4*)&As[arow][aseg * 16]     = tmp.q[0];
      *(u32x4*)&As[arow][aseg * 16 + 8] = tmp.q[1];
    }
    // ---- stage B tile 32x256 transposed into Bs[n][k], paired b32 stores ----
    {
      const float* bsrc = Bp + (long)k0 * ldB + n0 + tid;
#pragma unroll
      for (int kk = 0; kk < 32; kk += 2) {
        union { bf16 h[2]; unsigned u; } p;
        p.h[0] = (bf16)bsrc[(long)kk * ldB];
        p.h[1] = (bf16)bsrc[(long)(kk + 1) * ldB];
        *(unsigned*)&Bs[tid][kk] = p.u;
      }
    }
    __syncthreads();

    // A fragment: lane l16 -> row, lanes>=16 take K 8..15 / 24..31
    BFrag a;
    {
      const bf16* ar = &As[wave * 16 + l16][lh * 8];
      a.u4[0] = *(const u32x4*)(ar);
      a.u4[1] = *(const u32x4*)(ar + 16);
    }
#pragma unroll
    for (int j = 0; j < 16; ++j) {
      BFrag bb;
      const bf16* br = &Bs[j * 16 + l16][lh * 8];
      bb.u4[0] = *(const u32x4*)(br);
      bb.u4[1] = *(const u32x4*)(br + 16);
      acc[j] = wmma_bf16(a, bb, acc[j]);
    }
    __syncthreads();
  }

  const int wrow0 = m0 + wave * 16 + 8 * lh; // + r gives global row (C layout)

  if (ROPE) {
    // pair (i, i+128) with i = j*16 + l16 (n0 == 0 for rope-mode launches)
#pragma unroll
    for (int j = 0; j < 8; ++j) {
      int   i    = j * 16 + l16;
      float invf = __expf(-(float)(2 * i) * (LOG1E4 / (float)Hn));
#pragma unroll
      for (int r = 0; r < 8; ++r) {
        float fr = (float)(wrow0 + r) * invf;
        float sn, cs;
        __sincosf(fr, &sn, &cs);
        float x1 = acc[j][r], x2 = acc[j + 8][r];
        acc[j][r]     = x1 * cs - x2 * sn;
        acc[j + 8][r] = x2 * cs + x1 * sn;
      }
    }
  }

#pragma unroll
  for (int j = 0; j < 16; ++j) {
#pragma unroll
    for (int r = 0; r < 8; ++r) {
      int row = wrow0 + r;
      int col = n0 + j * 16 + l16;
      Cp[(long)row * ldC + col] = (CT)(acc[j][r] * outScale);
    }
  }
}

// ---------------------------------------------------------------------------
// Flash-style sliding-window attention with tanh soft-cap.
// q,k,v: bf16 [B][heads][S][H]; av out: bf16 [B][S][NQ][H].
// Block = 8 waves, wave owns 16 query rows; key blocks of 32 staged in LDS.
// K tile staged with CDNA5 async global->LDS copies (ASYNCcnt-tracked).
// ---------------------------------------------------------------------------
__global__ __launch_bounds__(256)
void attn_kernel(const bf16* __restrict__ q, const bf16* __restrict__ k,
                 const bf16* __restrict__ v, bf16* __restrict__ av)
{
  __shared__ bf16 Ks[32][Hn + 8];   // [key][h], row stride 528B (16B aligned)
  __shared__ bf16 Vs[Hn][40];       // transposed [h][key]
  __shared__ bf16 Ps[8][16][40];    // per-wave P tile [row][key]

  const int z    = blockIdx.z;                 // b*NKV*G + kv*G + g
  const int g    = z % Gn;
  const int kv   = (z / Gn) % NKVn;
  const int b    = z / (NKVn * Gn);
  const int n    = kv * Gn + g;
  const int tb   = blockIdx.y * 128;
  const int tid  = threadIdx.x;
  const int wave = tid >> 5;
  const int lane = tid & 31;
  const int l16  = lane & 15;
  const int lh   = lane >> 4;
  const int tw   = tb + wave * 16;

  const bf16* qp = q + ((long)b * NQn  + n)  * Sn * Hn;
  const bf16* kp = k + ((long)b * NKVn + kv) * Sn * Hn;
  const bf16* vp = v + ((long)b * NKVn + kv) * Sn * Hn;

  // generic LDS pointer: low 32 bits are the LDS byte offset (aperture rule)
  const unsigned ks_base = (unsigned)(uintptr_t)(&Ks[0][0]);

  // Preload Q fragments for this wave's 16 rows (8 K-chunks of 32 over H=256)
  BFrag qf[8];
  {
    const bf16* qrow = qp + (long)(tw + l16) * Hn;
#pragma unroll
    for (int hc = 0; hc < 8; ++hc) {
      const bf16* p = qrow + hc * 32 + lh * 8;
      qf[hc].u4[0] = *(const u32x4*)p;
      qf[hc].u4[1] = *(const u32x4*)(p + 16);
    }
  }

  v8f O[16];
#pragma unroll
  for (int j = 0; j < 16; ++j) O[j] = v8f_zero();
  float mrow[8], lrow[8];
#pragma unroll
  for (int r = 0; r < 8; ++r) { mrow[r] = -__builtin_inff(); lrow[r] = 0.f; }

  int kb0 = tb - (WIN - 1);
  if (kb0 < 0) kb0 = 0;
  kb0 &= ~31;

  for (int kb = kb0; kb < tb + 128; kb += 32) {
    // ---- stage K block [32][256] via async global->LDS b128 copies ----
#pragma unroll
    for (int i = 0; i < 4; ++i) {
      int c   = i * 256 + tid;          // 1024 chunks of 16B
      int r   = c >> 5;                 // key row 0..31
      int col = c & 31;                 // 16B unit within 512B row
      unsigned    lds = ks_base + (unsigned)(r * (Hn + 8) * 2 + col * 16);
      const bf16* gsrc = kp + (long)(kb + r) * Hn + col * 8;
      asm volatile("global_load_async_to_lds_b128 %0, %1, off"
                   :: "v"(lds), "v"(gsrc) : "memory");
    }
    // ---- stage V block transposed [256][32] ----
#pragma unroll
    for (int i = 0; i < 32; ++i) {
      int idx = i * 256 + tid;
      int r = idx >> 8, h = idx & 255;
      Vs[h][r] = vp[(long)(kb + r) * Hn + h];
    }
    asm volatile("s_wait_asynccnt 0x0" ::: "memory");
    __syncthreads();

    const bool active = (kb <= tw + 15) && (kb + 31 >= tw - (WIN - 1));
    if (active) {
      // ---- logits: two 16x16 tiles (keys kb..kb+15, kb+16..kb+31) ----
      v8f Sl[2];
#pragma unroll
      for (int j = 0; j < 2; ++j) {
        v8f s = v8f_zero();
#pragma unroll
        for (int hc = 0; hc < 8; ++hc) {
          BFrag kf;
          const bf16* p = &Ks[j * 16 + l16][hc * 32 + lh * 8];
          kf.u4[0] = *(const u32x4*)p;
          kf.u4[1] = *(const u32x4*)(p + 16);
          s = wmma_bf16(qf[hc], kf, s);
        }
        Sl[j] = s;
      }

      // ---- soft-cap + mask + online softmax (row stats per lane group) ----
      float sc[8];
#pragma unroll
      for (int r = 0; r < 8; ++r) {
        int   t  = tw + r + 8 * lh;
        int   s0 = kb + l16, s1 = s0 + 16;
        float a0 = CAP * fast_tanh(Sl[0][r] * (1.f / CAP));
        float a1 = CAP * fast_tanh(Sl[1][r] * (1.f / CAP));
        a0 = ((s0 <= t) && (s0 >= t - (WIN - 1))) ? a0 : -__builtin_inff();
        a1 = ((s1 <= t) && (s1 >= t - (WIN - 1))) ? a1 : -__builtin_inff();
        float mx = fmaxf(a0, a1);
#pragma unroll
        for (int off = 8; off; off >>= 1) mx = fmaxf(mx, __shfl_xor(mx, off));
        float mn   = fmaxf(mrow[r], mx);
        float safe = (mn == -__builtin_inff()) ? 0.f : mn;
        float scl  = __expf(mrow[r] - safe);
        float p0   = __expf(a0 - safe);
        float p1   = __expf(a1 - safe);
        float rs   = p0 + p1;
#pragma unroll
        for (int off = 8; off; off >>= 1) rs += __shfl_xor(rs, off);
        lrow[r] = lrow[r] * scl + rs;
        mrow[r] = mn;
        sc[r]   = scl;
        Ps[wave][r + 8 * lh][l16]      = (bf16)p0;
        Ps[wave][r + 8 * lh][16 + l16] = (bf16)p1;
      }
#pragma unroll
      for (int j = 0; j < 16; ++j)
#pragma unroll
        for (int r = 0; r < 8; ++r) O[j][r] *= sc[r];

      // ---- P (A-frag from per-wave LDS scratch) x V ----
      BFrag pf;
      {
        const bf16* pr = &Ps[wave][l16][lh * 8];
        pf.u4[0] = *(const u32x4*)pr;
        pf.u4[1] = *(const u32x4*)(pr + 16);
      }
#pragma unroll
      for (int j = 0; j < 16; ++j) {
        BFrag vf;
        const bf16* vr = &Vs[j * 16 + l16][lh * 8];
        vf.u4[0] = *(const u32x4*)vr;
        vf.u4[1] = *(const u32x4*)(vr + 16);
        O[j] = wmma_bf16(pf, vf, O[j]);
      }
    }
    __syncthreads();
  }

  // ---- normalize + store av[b][t][n][h] ----
  bf16* op = av + ((long)b * Sn) * (NQn * Hn) + (long)n * Hn;
#pragma unroll
  for (int j = 0; j < 16; ++j) {
#pragma unroll
    for (int r = 0; r < 8; ++r) {
      int t = tw + r + 8 * lh;
      int h = j * 16 + l16;
      op[(long)t * (NQn * Hn) + h] = (bf16)(O[j][r] / lrow[r]);
    }
  }
}

// ---------------------------------------------------------------------------
extern "C" void kernel_launch(void* const* d_in, const int* in_sizes, int n_in,
                              void* d_out, int out_size, void* d_ws, size_t ws_size,
                              hipStream_t stream)
{
  const float* x  = (const float*)d_in[0];   // [B,S,D]
  const float* Wq = (const float*)d_in[1];   // [NQ,D,H]
  const float* Wk = (const float*)d_in[2];   // [NKV,D,H]
  const float* Wv = (const float*)d_in[3];   // [NKV,D,H]
  const float* Wo = (const float*)d_in[4];   // [NQ,H,D]
  // d_in[5] = attention_mask (causal tril): implemented analytically.
  float* out = (float*)d_out;                // [B,S,D]

  // Workspace layout (bf16): q 32MB | k 16MB | v 16MB | av 32MB  (~96MB total)
  bf16* qb  = (bf16*)d_ws;
  bf16* kbf = qb  + (size_t)Bn * NQn  * Sn * Hn;
  bf16* vbf = kbf + (size_t)Bn * NKVn * Sn * Hn;
  bf16* avb = vbf + (size_t)Bn * NKVn * Sn * Hn;

  dim3 blk(256);

  // Q = rope(x . Wq) * 1/sqrt(H)   -- per (b, head) job
  gemm_kernel<false, bf16, true><<<dim3(1, Sn / 128, Bn * NQn), blk, 0, stream>>>(
      x, Wq, qb, Dn, Dn, Hn, Hn,
      NQn, (long)Sn * Dn, (long)Dn * Hn, (long)Sn * Hn, 1.0f / 16.0f);

  // K = rope(x . Wk)
  gemm_kernel<false, bf16, true><<<dim3(1, Sn / 128, Bn * NKVn), blk, 0, stream>>>(
      x, Wk, kbf, Dn, Dn, Hn, Hn,
      NKVn, (long)Sn * Dn, (long)Dn * Hn, (long)Sn * Hn, 1.0f);

  // V = x . Wv
  gemm_kernel<false, bf16, false><<<dim3(1, Sn / 128, Bn * NKVn), blk, 0, stream>>>(
      x, Wv, vbf, Dn, Dn, Hn, Hn,
      NKVn, (long)Sn * Dn, (long)Dn * Hn, (long)Sn * Hn, 1.0f);

  // Windowed flash attention -> av [B][S][NQ][H]
  attn_kernel<<<dim3(1, Sn / 128, Bn * NKVn * Gn), blk, 0, stream>>>(qb, kbf, vbf, avb);

  // out = av . Wo  as one GEMM per batch: [S, NQ*H] x [NQ*H, D]
  gemm_kernel<true, float, false><<<dim3(Dn / 256, Sn / 128, Bn), blk, 0, stream>>>(
      avb, Wo, out, NQn * Hn, NQn * Hn, Dn, Dn,
      1, (long)Sn * NQn * Hn, 0L, (long)Sn * Dn, 1.0f);
}
